// QResonanceBank_52355651338329
// MI455X (gfx1250) — matrix operand order repaired and checked by reading, not compiled, not run on hardware
//
#include <hip/hip_runtime.h>

// ---------------------------------------------------------------------------
// QResonanceBank on MI455X (gfx1250, wave32).
// Pipeline:
//   K1 pool+absmax   : traces[256,64,1024] -> pooled[256,64,128], invscale[256]
//   K2 pack A (f16)  : Ah[b][t][c]  (M=(b,t)=32768 rows, K=c=64)  normalized
//   K2b pack W (f16) : Wh[r][c]     (B matrix, K=c, N=r)
//   K3 fused         : per 16b x 16r tile: 128 x (16x16 drive tiles) via
//                      v_wmma_f32_16x16x32_f16 (bias in C), drive kept in
//                      128KB LDS, then 128-step resonator recurrence -> out.
// ---------------------------------------------------------------------------

typedef __attribute__((ext_vector_type(16))) _Float16 v16h;
typedef __attribute__((ext_vector_type(8)))  _Float16 v8h;
typedef __attribute__((ext_vector_type(8)))  float    v8f;

#define B_N   256
#define C_N   64
#define T_N   1024
#define R_N   512
#define TB_N  128   // TIME_BINS

// ---------------- K1: adaptive_avg_pool1d(1024->128) + per-sample absmax ----
__global__ __launch_bounds__(256) void pool_absmax_kernel(
    const float* __restrict__ traces,   // [B][C][T]
    float* __restrict__ pooled,         // [B][C][TB]
    float* __restrict__ invscale)       // [B]
{
    const int b   = blockIdx.x;
    const int tid = threadIdx.x;
    const float* tb = traces + (size_t)b * (C_N * T_N);
    float*       pb = pooled + (size_t)b * (C_N * TB_N);

    float amax = 0.0f;
    #pragma unroll
    for (int i = 0; i < 32; ++i) {
        const int e = tid + i * 256;            // e = c*128 + t  (0..8191)
        const float* src = tb + (size_t)e * 8;  // c*1024 + t*8 == e*8 : contiguous
        float4 x0 = *(const float4*)(src);
        float4 x1 = *(const float4*)(src + 4);
        float s = ((x0.x + x0.y) + (x0.z + x0.w)) + ((x1.x + x1.y) + (x1.z + x1.w));
        float m = s * 0.125f;
        pb[e] = m;
        amax = fmaxf(amax, fabsf(m));
    }
    __shared__ float red[256];
    red[tid] = amax;
    __syncthreads();
    for (int s = 128; s > 0; s >>= 1) {
        if (tid < s) red[tid] = fmaxf(red[tid], red[tid + s]);
        __syncthreads();
    }
    if (tid == 0) invscale[b] = 1.0f / fmaxf(red[0], 1.0f);
}

// ---------------- K2: pack normalized pooled -> f16 A matrix [b][t][c] ------
__global__ __launch_bounds__(256) void pack_a_kernel(
    const float* __restrict__ pooled,     // [B][C][TB]
    const float* __restrict__ invscale,   // [B]
    _Float16* __restrict__ Ah)            // [B][TB][C]
{
    const int e   = blockIdx.x * 256 + threadIdx.x;   // < B*C*TB
    const int b   = e >> 13;          // / 8192
    const int rem = e & 8191;
    const int c   = rem >> 7;         // / 128
    const int t   = rem & 127;
    const float v = pooled[e] * invscale[b];
    Ah[((size_t)b * TB_N + t) * C_N + c] = (_Float16)v;
}

// ---------------- K2b: W fp32 -> f16, same [r][c] layout --------------------
__global__ __launch_bounds__(256) void pack_w_kernel(
    const float* __restrict__ W, _Float16* __restrict__ Wh)
{
    const int e = blockIdx.x * 256 + threadIdx.x;     // < R*C
    Wh[e] = (_Float16)W[e];
}

// A-fragment: lane<16 holds K {k0..k0+15} (contiguous); row base `p` points
// at Ah[b][t][0]; koff already includes the lane-half offset.
static __device__ inline v16h load_afrag(const _Float16* __restrict__ p, int koff) {
    v8h lo = *(const v8h*)(p + koff);
    v8h hi = *(const v8h*)(p + koff + 8);
    return __builtin_shufflevector(lo, hi, 0,1,2,3,4,5,6,7,8,9,10,11,12,13,14,15);
}

static __device__ inline float sigmoidf_dev(float x) {
    return 1.0f / (1.0f + expf(-x));
}
static __device__ inline float softplusf_dev(float x) {
    return (x > 20.0f) ? x : log1pf(expf(x));
}

// ---------------- K3: fused WMMA GEMM + resonator recurrence ---------------
__global__ __launch_bounds__(256) void gemm_resonate_kernel(
    const _Float16* __restrict__ Ah,      // [B][TB][C]
    const _Float16* __restrict__ Wh,      // [R][C]
    const float* __restrict__ bias,       // [R]
    const float* __restrict__ freq_raw,   // [R]
    const float* __restrict__ q_raw,      // [R]
    const float* __restrict__ thr_raw,    // [R]
    float* __restrict__ out)              // [B][R]
{
    extern __shared__ float sdrive[];     // [TB][16b][16r] = 128*256 floats = 128KB

    const int r0   = blockIdx.x * 16;
    const int b0   = blockIdx.y * 16;
    const int tid  = threadIdx.x;         // 0..255, 8 waves
    const int wave = tid >> 5;
    const int lane = tid & 31;
    const int lh   = lane & 15;
    const int hiH  = lane >> 4;           // 0 / 1

    // ---- Phase 1: drive = Ah x Wh^T + bias, one 16x16 tile per time step ----
    // B fragments (shared across the 16 time steps this wave owns):
    // lane n holds column r0+n; lo half K 0..15, hi half K 16..31 (contiguous).
    const _Float16* wrow = Wh + (size_t)(r0 + lh) * C_N + hiH * 16;
    const v16h bfrag0 = *(const v16h*)(wrow);        // K  0..31 block
    const v16h bfrag1 = *(const v16h*)(wrow + 32);   // K 32..63 block

    const float bv = bias[r0 + lh];                  // folded into C accumulator

    const int akoff = hiH * 16;                      // lane-half K offset for A

    for (int j = 0; j < 16; ++j) {
        const int t = wave * 16 + j;
        const _Float16* arow = Ah + ((size_t)(b0 + lh) * TB_N + t) * C_N;
        v16h a0 = load_afrag(arow, akoff);           // K  0..31
        v16h a1 = load_afrag(arow, akoff + 32);      // K 32..63

        v8f acc = { bv, bv, bv, bv, bv, bv, bv, bv };
        acc = __builtin_amdgcn_wmma_f32_16x16x32_f16(
                  false, a0, false, bfrag0, (short)0, acc, false, false);
        acc = __builtin_amdgcn_wmma_f32_16x16x32_f16(
                  false, a1, false, bfrag1, (short)0, acc, false, false);

        // D layout: VGPR i -> M = i (lanes 0-15) / 8+i (lanes 16-31), N = lh.
        #pragma unroll
        for (int i = 0; i < 8; ++i) {
            const int brow = hiH * 8 + i;
            sdrive[t * 256 + brow * 16 + lh] = acc[i];
        }
    }
    __syncthreads();

    // ---- Phase 2: 128-step resonate-and-fire per (b,r); 1 thread each ------
    const int bi = tid >> 4;              // 0..15
    const int ri = tid & 15;              // 0..15
    const int r  = r0 + ri;

    const float freq  = 0.03f + 0.17f * sigmoidf_dev(freq_raw[r]);
    const float qf    = 1.5f + softplusf_dev(q_raw[r]);
    const float decay = expf(-1.0f / qf);
    const float thr   = 0.35f + 0.75f * sigmoidf_dev(thr_raw[r]);

    float state = 0.0f, vel = 0.0f, cnt = 0.0f;
    for (int t = 0; t < TB_N; ++t) {
        const float cur = sdrive[t * 256 + tid];   // tid == bi*16+ri: conflict-free
        vel   = decay * vel + cur - freq * state;
        state = state + freq * vel;
        const float sp = (state > thr) ? 1.0f : 0.0f;
        cnt  += sp;
        state -= sp * thr;
    }
    out[(size_t)(b0 + bi) * R_N + r] = cnt * (1.0f / (float)TB_N);
}

// ---------------------------------------------------------------------------
extern "C" void kernel_launch(void* const* d_in, const int* in_sizes, int n_in,
                              void* d_out, int out_size, void* d_ws, size_t ws_size,
                              hipStream_t stream) {
    const float* traces   = (const float*)d_in[0];   // [256,64,1024]
    const float* W        = (const float*)d_in[1];   // [512,64]
    const float* bias     = (const float*)d_in[2];   // [512]
    const float* freq_raw = (const float*)d_in[3];   // [512]
    const float* q_raw    = (const float*)d_in[4];   // [512]
    const float* thr_raw  = (const float*)d_in[5];   // [512]
    float* out = (float*)d_out;                      // [256,512]

    // Workspace layout (all 32B-aligned): pooled f32 | invscale | Ah f16 | Wh f16
    float*    pooled   = (float*)d_ws;                         // 2,097,152 f32
    float*    invscale = pooled + (size_t)B_N * C_N * TB_N;    // 256 f32
    _Float16* Ah       = (_Float16*)(invscale + 256);          // 2,097,152 f16
    _Float16* Wh       = Ah + (size_t)B_N * TB_N * C_N;        // 32,768 f16

    pool_absmax_kernel<<<B_N, 256, 0, stream>>>(traces, pooled, invscale);
    pack_a_kernel<<<(B_N * C_N * TB_N) / 256, 256, 0, stream>>>(pooled, invscale, Ah);
    pack_w_kernel<<<(R_N * C_N) / 256, 256, 0, stream>>>(W, Wh);

    const int smem = TB_N * 16 * 16 * (int)sizeof(float);      // 128 KB (<320KB WGP LDS)
    (void)hipFuncSetAttribute(reinterpret_cast<const void*>(gemm_resonate_kernel),
                              hipFuncAttributeMaxDynamicSharedMemorySize, smem);
    gemm_resonate_kernel<<<dim3(R_N / 16, B_N / 16), 256, smem, stream>>>(
        Ah, Wh, bias, freq_raw, q_raw, thr_raw, out);
}
